// SliceAttention_78898549228061
// MI455X (gfx1250) — matrix operand order repaired and checked by reading, not compiled, not run on hardware
//
#include <hip/hip_runtime.h>
#include <hip/hip_bf16.h>
#include <math.h>

#define NROWS 262144
#define CDIM  512
#define ADIM  128
#define SSEG  1024

typedef __attribute__((ext_vector_type(16))) __bf16         v16bf;
typedef __attribute__((ext_vector_type(16))) float          v16f;
typedef __attribute__((ext_vector_type(8)))  float          v8f;

__device__ __forceinline__ unsigned short f2bf(float f) {
    unsigned int u = __float_as_uint(f);
    u += 0x7fffu + ((u >> 16) & 1u);          // round-to-nearest-even
    return (unsigned short)(u >> 16);
}

// ---------------------------------------------------------------------------
// Kernel 0: convert w1,w2 (fp32 [512][128]) to bf16, pre-swizzled into WMMA
// B-fragment order. Fragment f = kb*8+nt (kb: K-block of 32, nt: 16-col tile).
// Within a fragment: lane l holds column nt*16+(l&15); its 16 bf16 values are
// K = kb*32 + (l>=16 ? 8 : 0) + (p&7) + (p>=8 ? 16 : 0), p=0..15 (matches 32x16 bf16 B layout).
// ---------------------------------------------------------------------------
__global__ void prep_weights(const float* __restrict__ w1,
                             const float* __restrict__ w2,
                             unsigned short* __restrict__ wsw) {
    int i   = blockIdx.x * 256 + threadIdx.x;   // 0..131071 (2 matrices)
    int mat = i >> 16;
    int e   = i & 65535;
    int f   = e >> 9;          // fragment 0..127
    int r   = e & 511;
    int l   = r >> 4;          // lane 0..31
    int p   = r & 15;          // K position within lane
    int kb  = f >> 3, nt = f & 7;
    int K   = kb * 32 + ((l >= 16) ? 8 : 0) + (p & 7) + ((p >= 8) ? 16 : 0);
    int Nc  = nt * 16 + (l & 15);
    const float* w = mat ? w2 : w1;
    wsw[i] = f2bf(w[K * ADIM + Nc]);
}

// ---------------------------------------------------------------------------
// Kernel 1: gated attention score via WMMA bf16.
// 256 threads = 8 waves; wave w owns rows [blk*128 + w*16, +16).
// Both swizzled weight matrices staged in LDS (256 KB dynamic) via the
// CDNA5 async-to-LDS path (ASYNCcnt). B fragments use parity-alternating
// double buffers (non-coalescable) so ds_load latency overlaps the WMMAs;
// global x loads for the next K-step are pipelined across the whole K-body.
// ---------------------------------------------------------------------------
__global__ __launch_bounds__(256, 1) void score_kernel(
    const float* __restrict__ x,
    const unsigned short* __restrict__ wsw,     // 131072 bf16 (w1 | w2)
    const float* __restrict__ b1,  const float* __restrict__ b2,
    const float* __restrict__ w_score, const float* __restrict__ b_score,
    float* __restrict__ score)
{
    extern __shared__ unsigned short wlds[];    // 131072 ushort = 256 KB
    const int t    = threadIdx.x;
    const int lane = t & 31;
    const int wv   = t >> 5;

    // Async staging of both swizzled weight matrices into LDS.
    // Each lane moves 16B per issue; 256 threads x 64 iters x 16B = 256 KB.
    {
        unsigned      ldsbase = (unsigned)(unsigned long long)(void*)wlds;
        unsigned long long g  = (unsigned long long)(const void*)wsw;
        for (int i = 0; i < 64; ++i) {
            unsigned boff = (unsigned)(t * 16 + i * 4096);
            unsigned ldst = ldsbase + boff;
            asm volatile("global_load_async_to_lds_b128 %0, %1, %2"
                         :: "v"(ldst), "v"(boff), "s"(g)
                         : "memory");
        }
        asm volatile("s_wait_asynccnt 0" ::: "memory");
    }
    __syncthreads();

    const int    rowbase = blockIdx.x * 128 + wv * 16;
    const int    mrow    = rowbase + (lane & 15);
    const float* xr      = x + (size_t)mrow * CDIM;
    const int    koff    = (lane >> 4) ? 8 : 0;   // A-layout K half select

    v8f zero = {};
    v8f accV[8], accU[8];
    #pragma unroll
    for (int nt = 0; nt < 8; ++nt) { accV[nt] = zero; accU[nt] = zero; }

    // Preload x for kb = 0.
    const float4* xp0 = (const float4*)(xr + koff);
    float4 f0 = xp0[0], f1 = xp0[1];   // K = base+0..7
    float4 f2 = xp0[4], f3 = xp0[5];   // K = base+16..23

    #pragma unroll 1
    for (int kb = 0; kb < 16; ++kb) {
        // Convert current A fragment (16x32 bf16).
        v16f xf;
        xf[0]  = f0.x; xf[1]  = f0.y; xf[2]  = f0.z; xf[3]  = f0.w;
        xf[4]  = f1.x; xf[5]  = f1.y; xf[6]  = f1.z; xf[7]  = f1.w;
        xf[8]  = f2.x; xf[9]  = f2.y; xf[10] = f2.z; xf[11] = f2.w;
        xf[12] = f3.x; xf[13] = f3.y; xf[14] = f3.z; xf[15] = f3.w;
        v16bf afrag = __builtin_convertvector(xf, v16bf);   // packed cvt

        // Pipeline the next K-step's x loads across the WMMA body.
        if (kb < 15) {
            const float4* xn = (const float4*)(xr + (kb + 1) * 32 + koff);
            f0 = xn[0]; f1 = xn[1]; f2 = xn[4]; f3 = xn[5];
        }

        // Parity-alternating double-buffered B fragments.
        const unsigned short* wp = wlds + kb * 4096 + lane * 16;
        v16bf bVa = *(const v16bf*)(wp);
        v16bf bUa = *(const v16bf*)(wp + 65536);
        v16bf bVb = *(const v16bf*)(wp + 512);
        v16bf bUb = *(const v16bf*)(wp + 65536 + 512);
        #pragma unroll
        for (int nt = 0; nt < 8; nt += 2) {
            v16bf cV = bVa, cU = bUa;
            if (nt + 2 < 8) {                 // prefetch nt+2 into buffer A
                bVa = *(const v16bf*)(wp + (nt + 2) * 512);
                bUa = *(const v16bf*)(wp + 65536 + (nt + 2) * 512);
            }
            accV[nt] = __builtin_amdgcn_wmma_f32_16x16x32_bf16(
                false, afrag, false, cV, (short)0, accV[nt], false, false);
            accU[nt] = __builtin_amdgcn_wmma_f32_16x16x32_bf16(
                false, afrag, false, cU, (short)0, accU[nt], false, false);

            v16bf dV = bVb, dU = bUb;
            if (nt + 3 < 8) {                 // prefetch nt+3 into buffer B
                bVb = *(const v16bf*)(wp + (nt + 3) * 512);
                bUb = *(const v16bf*)(wp + 65536 + (nt + 3) * 512);
            }
            accV[nt + 1] = __builtin_amdgcn_wmma_f32_16x16x32_bf16(
                false, afrag, false, dV, (short)0, accV[nt + 1], false, false);
            accU[nt + 1] = __builtin_amdgcn_wmma_f32_16x16x32_bf16(
                false, afrag, false, dU, (short)0, accU[nt + 1], false, false);
        }
    }

    // Fused epilogue: score_m = b_score + sum_a tanh(V)*sigmoid(U)*w_score[a].
    // C layout: lane holds col a = nt*16+(lane&15); vgpr k -> row k + 8*(lane>=16).
    float part[8];
    #pragma unroll
    for (int k = 0; k < 8; ++k) part[k] = 0.f;
    #pragma unroll
    for (int nt = 0; nt < 8; ++nt) {
        int   a   = nt * 16 + (lane & 15);
        float bb1 = b1[a], bb2 = b2[a], ws = w_score[a];
        #pragma unroll
        for (int k = 0; k < 8; ++k) {
            float zv = accV[nt][k] + bb1;
            float e2 = __expf(2.0f * zv);
            float v  = (e2 - 1.0f) / (e2 + 1.0f);           // tanh
            float u  = 1.0f / (1.0f + __expf(-(accU[nt][k] + bb2)));
            part[k] += v * u * ws;
        }
    }
    // Reduce over the 16-lane N-group (wave32 shuffles).
    #pragma unroll
    for (int k = 0; k < 8; ++k) {
        #pragma unroll
        for (int m = 1; m < 16; m <<= 1) part[k] += __shfl_xor(part[k], m, 32);
    }
    if ((lane & 15) == 0) {
        float bs = b_score[0];
        int   mb = (lane >> 4) * 8;
        #pragma unroll
        for (int k = 0; k < 8; ++k) score[rowbase + mb + k] = part[k] + bs;
    }
}

// ---------------------------------------------------------------------------
// Kernel 2: per-segment stable softmax + weighted pooling + MLP head.
// One 256-thread block per segment; thread t owns channels t and t+256.
// ---------------------------------------------------------------------------
__global__ __launch_bounds__(256) void pool_kernel(
    const float* __restrict__ x,   const int* __restrict__ idx,
    const float* __restrict__ score,
    const float* __restrict__ w_m1, const float* __restrict__ b_m1,
    const float* __restrict__ w_m2, const float* __restrict__ b_m2,
    float* __restrict__ pred)
{
    __shared__ float red[256];
    __shared__ float pooled[512];
    __shared__ float hbuf[64];
    const int s     = blockIdx.x;
    const int t     = threadIdx.x;
    const int start = (s == 0) ? 0 : idx[s - 1];
    const int end   = idx[s];

    // Pass 1: segment max
    float m = -INFINITY;
    for (int r = start + t; r < end; r += 256) m = fmaxf(m, score[r]);
    red[t] = m; __syncthreads();
    for (int o = 128; o > 0; o >>= 1) {
        if (t < o) red[t] = fmaxf(red[t], red[t + o]);
        __syncthreads();
    }
    m = red[0]; __syncthreads();

    // Pass 2: sum of exp
    float sum = 0.f;
    for (int r = start + t; r < end; r += 256) sum += __expf(score[r] - m);
    red[t] = sum; __syncthreads();
    for (int o = 128; o > 0; o >>= 1) {
        if (t < o) red[t] += red[t + o];
        __syncthreads();
    }
    float denom = red[0]; __syncthreads();
    float inv = (denom > 0.f) ? 1.f / denom : 0.f;

    // Pass 3: weighted-sum pooling (coalesced row reads, prefetch ahead)
    float a0 = 0.f, a1 = 0.f;
    for (int r = start; r < end; ++r) {
        float        w  = __expf(score[r] - m) * inv;
        const float* xr = x + (size_t)r * CDIM;
        __builtin_prefetch(xr + 4 * CDIM + t, 0, 1);
        a0 += w * xr[t];
        a1 += w * xr[t + 256];
    }
    pooled[t] = a0; pooled[t + 256] = a1;
    __syncthreads();

    // MLP head: h = relu(pooled @ w_m1 + b_m1); pred = relu(h @ w_m2 + b_m2)
    if (t < 64) {
        float h = b_m1[t];
        #pragma unroll 8
        for (int c = 0; c < 512; ++c) h += pooled[c] * w_m1[c * 64 + t];
        hbuf[t] = fmaxf(h, 0.f);
    }
    __syncthreads();
    if (t < 32) {
        float p = hbuf[t] * w_m2[t] + hbuf[t + 32] * w_m2[t + 32];
        #pragma unroll
        for (int o = 16; o > 0; o >>= 1) p += __shfl_xor(p, o, 32);
        if (t == 0) pred[s] = fmaxf(p + b_m2[0], 0.f);
    }
}

// ---------------------------------------------------------------------------
extern "C" void kernel_launch(void* const* d_in, const int* in_sizes, int n_in,
                              void* d_out, int out_size, void* d_ws, size_t ws_size,
                              hipStream_t stream) {
    const float* x       = (const float*)d_in[0];
    const int*   idx     = (const int*)  d_in[1];
    const float* w1      = (const float*)d_in[2];
    const float* b1      = (const float*)d_in[3];
    const float* w2      = (const float*)d_in[4];
    const float* b2      = (const float*)d_in[5];
    const float* w_score = (const float*)d_in[6];
    const float* b_score = (const float*)d_in[7];
    const float* w_m1    = (const float*)d_in[8];
    const float* b_m1    = (const float*)d_in[9];
    const float* w_m2    = (const float*)d_in[10];
    const float* b_m2    = (const float*)d_in[11];
    float* pred = (float*)d_out;

    // Workspace: [0,256KB) swizzled bf16 weights, then N fp32 scores.
    unsigned short* wsw   = (unsigned short*)d_ws;
    float*          score = (float*)((char*)d_ws + 262144);

    (void)in_sizes; (void)n_in; (void)out_size; (void)ws_size;

    hipFuncSetAttribute(reinterpret_cast<const void*>(score_kernel),
                        hipFuncAttributeMaxDynamicSharedMemorySize, 262144);

    prep_weights<<<512, 256, 0, stream>>>(w1, w2, wsw);
    score_kernel<<<NROWS / 128, 256, 262144, stream>>>(x, wsw, b1, b2,
                                                       w_score, b_score, score);
    pool_kernel<<<SSEG, 256, 0, stream>>>(x, idx, score, w_m1, b_m1,
                                          w_m2, b_m2, pred);
}
